// ProtoTSNet_64579128263011
// MI455X (gfx1250) — matrix operand order repaired and checked by reading, not compiled, not run on hardware
//
#include <hip/hip_runtime.h>
#include <math.h>

// ---------------- problem constants ----------------
#define NB      64      // batch
#define FIN     32      // input channels
#define TT      2048    // time length
#define TP      2096    // padded xT rows (covers L1P-1 + ENC_K-1 = 2094)
#define LAT     128     // latent channels
#define ENCK    16      // encoder kernel width
#define KENC    512     // encoder GEMM K = FIN*ENCK
#define L1P     2080    // padded h rows (valid: 2033; covers 2047 + 31 = 2078)
#define NP      200     // prototypes
#define PP      256     // padded prototypes (4 * 64)
#define PL      32      // prototype length
#define KPROTO  4096    // proto GEMM K = LAT*PL
#define LOUT    2002    // valid output columns
#define X2PSN   2048    // padded x2ps columns (32 N-blocks of 64)
#define NCLS    10
#define EPS_    1e-4f

typedef __attribute__((ext_vector_type(16))) __bf16 bf16x16;
typedef __attribute__((ext_vector_type(8)))  __bf16 bf16x8;
typedef __attribute__((ext_vector_type(8)))  float  f32x8;
typedef __attribute__((ext_vector_type(8)))  unsigned short u16x8;

union AB16 { bf16x16 v; bf16x8 h[2]; };

__device__ __forceinline__ unsigned short f2bf(float f) {
  unsigned int u = __float_as_uint(f);
  unsigned int r = u + 0x7FFFu + ((u >> 16) & 1u);   // round to nearest even
  return (unsigned short)(r >> 16);
}
__device__ __forceinline__ float bf2f(unsigned short s) {
  return __uint_as_float(((unsigned int)s) << 16);
}

#define WMMA_BF16(A, B, C) \
  __builtin_amdgcn_wmma_f32_16x16x32_bf16(false, (A), false, (B), (short)0, (C), false, false)

// ---------------- packing kernels ----------------

// xT[b][t][c] (bf16, c contiguous), zero for t >= TT
__global__ __launch_bounds__(256) void pack_xT(const float* __restrict__ x,
                                               unsigned short* __restrict__ xT) {
  int idx = blockIdx.x * blockDim.x + threadIdx.x;
  int total = NB * TP * FIN;
  if (idx >= total) return;
  int c = idx % FIN;
  int t = (idx / FIN) % TP;
  int b = idx / (FIN * TP);
  float v = (t < TT) ? x[((size_t)b * FIN + c) * TT + t] : 0.0f;
  xT[idx] = f2bf(v);
}

// encA[o][K] with K = kk*32 + c
__global__ __launch_bounds__(256) void pack_encA(const float* __restrict__ w,
                                                 unsigned short* __restrict__ A) {
  int idx = blockIdx.x * blockDim.x + threadIdx.x;
  if (idx >= LAT * KENC) return;
  int K = idx % KENC;
  int o = idx / KENC;
  int kk = K / FIN, c = K % FIN;
  A[idx] = f2bf(w[((size_t)o * FIN + c) * ENCK + kk]);
}

// addonA[o][c]
__global__ __launch_bounds__(256) void pack_addonA(const float* __restrict__ w,
                                                   unsigned short* __restrict__ A) {
  int idx = blockIdx.x * blockDim.x + threadIdx.x;
  if (idx >= LAT * LAT) return;
  A[idx] = f2bf(w[idx]);   // addon_w is [LAT][LAT][1] flat
}

// protoA[p][K] with K = kk*128 + c, zero-padded rows p >= NP
__global__ __launch_bounds__(256) void pack_protoA(const float* __restrict__ pr,
                                                   unsigned short* __restrict__ A) {
  int idx = blockIdx.x * blockDim.x + threadIdx.x;
  if (idx >= PP * KPROTO) return;
  int K = idx % KPROTO;
  int p = idx / KPROTO;
  int kk = K / LAT, c = K % LAT;
  float v = (p < NP) ? pr[((size_t)p * LAT + c) * PL + kk] : 0.0f;
  A[idx] = f2bf(v);
}

// p2[p] = sum_K protoA[p][K]^2  (from bf16-rounded values for consistency)
__global__ __launch_bounds__(256) void p2_kernel(const unsigned short* __restrict__ A,
                                                 float* __restrict__ p2) {
  int p = blockIdx.x * blockDim.x + threadIdx.x;
  if (p >= PP) return;
  float s = 0.0f;
  const unsigned short* row = A + (size_t)p * KPROTO;
  for (int k = 0; k < KPROTO; ++k) { float f = bf2f(row[k]); s += f * f; }
  p2[p] = s;
}

__global__ __launch_bounds__(256) void init_mind(unsigned int* __restrict__ mind) {
  int idx = blockIdx.x * blockDim.x + threadIdx.x;
  if (idx < NB * PP) mind[idx] = 0x7F800000u;  // +inf
}

// ---------------- GEMM epilogue helper ----------------
__device__ __forceinline__ void store_relu_bf16(unsigned short* __restrict__ h,
                                                int bb, int l, int pm, const f32x8& c) {
  u16x8 o;
#pragma unroll
  for (int v = 0; v < 8; ++v) {
    float f = c[v] > 0.0f ? c[v] : 0.0f;
    o[v] = f2bf(f);
  }
  *(u16x8*)(h + ((size_t)bb * L1P + l) * LAT + pm) = o;
}

// ---------------- encoder GEMM: h1 = relu(encA[128x512] @ im2col(xT)) ----------------
// waves: bb(64) x nblk(65) x mblk(4); 32x32 tile per wave
__global__ __launch_bounds__(256) void enc_gemm(const unsigned short* __restrict__ encA,
                                                const unsigned short* __restrict__ xT,
                                                unsigned short* __restrict__ h1) {
  int wid  = (blockIdx.x * blockDim.x + threadIdx.x) >> 5;
  int lane = threadIdx.x & 31;
  int mblk = wid & 3;            // 4 M-blocks of 32 (LAT=128)
  int rest = wid >> 2;
  int nblk = rest % 65;          // 65 N-blocks of 32 (L1P=2080)
  int bb   = rest / 65;
  int m0 = mblk * 32, l0 = nblk * 32;
  int m = lane & 15, hi = lane >> 4;

  f32x8 c00 = {0,0,0,0,0,0,0,0}, c01 = {0,0,0,0,0,0,0,0};
  f32x8 c10 = {0,0,0,0,0,0,0,0}, c11 = {0,0,0,0,0,0,0,0};
  const unsigned short* ar0 = encA + (size_t)(m0 + m) * KENC;
  const unsigned short* ar1 = ar0 + 16 * KENC;

#pragma unroll 4
  for (int kk = 0; kk < ENCK; ++kk) {
    AB16 a0, a1;
    a0.h[0] = *(const bf16x8*)(ar0 + kk * 32 + 16 * hi);
    a0.h[1] = *(const bf16x8*)(ar0 + kk * 32 + 8 + 16 * hi);
    a1.h[0] = *(const bf16x8*)(ar1 + kk * 32 + 16 * hi);
    a1.h[1] = *(const bf16x8*)(ar1 + kk * 32 + 8 + 16 * hi);
    const unsigned short* bp =
        xT + ((size_t)bb * TP + (l0 + m + kk)) * FIN + 16 * hi;
    bf16x16 b0 = *(const bf16x16*)(bp);
    bf16x16 b1 = *(const bf16x16*)(bp + 16 * FIN);
    c00 = WMMA_BF16(a0.v, b0, c00);
    c10 = WMMA_BF16(a1.v, b0, c10);
    c01 = WMMA_BF16(a0.v, b1, c01);
    c11 = WMMA_BF16(a1.v, b1, c11);
  }
  store_relu_bf16(h1, bb, l0 + m,      m0 + 8 * hi,      c00);
  store_relu_bf16(h1, bb, l0 + m,      m0 + 16 + 8 * hi, c10);
  store_relu_bf16(h1, bb, l0 + 16 + m, m0 + 8 * hi,      c01);
  store_relu_bf16(h1, bb, l0 + 16 + m, m0 + 16 + 8 * hi, c11);
}

// ---------------- addon 1x1 GEMM: h2 = relu(addonA[128x128] @ h1) ----------------
__global__ __launch_bounds__(256) void addon_gemm(const unsigned short* __restrict__ addA,
                                                  const unsigned short* __restrict__ h1,
                                                  unsigned short* __restrict__ h2) {
  int wid  = (blockIdx.x * blockDim.x + threadIdx.x) >> 5;
  int lane = threadIdx.x & 31;
  int mblk = wid & 3;
  int rest = wid >> 2;
  int nblk = rest % 65;
  int bb   = rest / 65;
  int m0 = mblk * 32, l0 = nblk * 32;
  int m = lane & 15, hi = lane >> 4;

  f32x8 c00 = {0,0,0,0,0,0,0,0}, c01 = {0,0,0,0,0,0,0,0};
  f32x8 c10 = {0,0,0,0,0,0,0,0}, c11 = {0,0,0,0,0,0,0,0};
  const unsigned short* ar0 = addA + (size_t)(m0 + m) * LAT;
  const unsigned short* ar1 = ar0 + 16 * LAT;
  const unsigned short* brow = h1 + ((size_t)bb * L1P + l0 + m) * LAT + 16 * hi;

#pragma unroll
  for (int kt = 0; kt < 4; ++kt) {
    AB16 a0, a1;
    a0.h[0] = *(const bf16x8*)(ar0 + kt * 32 + 16 * hi);
    a0.h[1] = *(const bf16x8*)(ar0 + kt * 32 + 8 + 16 * hi);
    a1.h[0] = *(const bf16x8*)(ar1 + kt * 32 + 16 * hi);
    a1.h[1] = *(const bf16x8*)(ar1 + kt * 32 + 8 + 16 * hi);
    bf16x16 b0 = *(const bf16x16*)(brow + kt * 32);
    bf16x16 b1 = *(const bf16x16*)(brow + kt * 32 + 16 * LAT);
    c00 = WMMA_BF16(a0.v, b0, c00);
    c10 = WMMA_BF16(a1.v, b0, c10);
    c01 = WMMA_BF16(a0.v, b1, c01);
    c11 = WMMA_BF16(a1.v, b1, c11);
  }
  store_relu_bf16(h2, bb, l0 + m,      m0 + 8 * hi,      c00);
  store_relu_bf16(h2, bb, l0 + m,      m0 + 16 + 8 * hi, c10);
  store_relu_bf16(h2, bb, l0 + 16 + m, m0 + 8 * hi,      c01);
  store_relu_bf16(h2, bb, l0 + 16 + m, m0 + 16 + 8 * hi, c11);
}

// ---------------- x2s / sliding window ----------------
__global__ __launch_bounds__(256) void x2s_kernel(const unsigned short* __restrict__ h2,
                                                  float* __restrict__ x2s) {
  int idx = blockIdx.x * blockDim.x + threadIdx.x;   // b*L1P + l
  if (idx >= NB * L1P) return;
  const unsigned short* row = h2 + (size_t)idx * LAT;
  float s = 0.0f;
#pragma unroll 8
  for (int c = 0; c < LAT; ++c) { float f = bf2f(row[c]); s += f * f; }
  x2s[idx] = s;
}

__global__ __launch_bounds__(256) void x2ps_kernel(const float* __restrict__ x2s,
                                                   float* __restrict__ x2ps) {
  int idx = blockIdx.x * blockDim.x + threadIdx.x;   // b*X2PSN + lo
  if (idx >= NB * X2PSN) return;
  int lo = idx % X2PSN;
  int b  = idx / X2PSN;
  const float* row = x2s + (size_t)b * L1P + lo;     // lo+31 <= 2078 < L1P
  float s = 0.0f;
#pragma unroll 8
  for (int k = 0; k < PL; ++k) s += row[k];
  x2ps[idx] = s;
}

// ---------------- prototype GEMM + fused distance/min ----------------
// 64x64 register tile per wave: 4x4 WMMA quadrants, 16 WMMAs / K-step,
// 12 fragment loads / K-step -> 32 FLOP per L2 byte.
__global__ __launch_bounds__(256) void proto_gemm(const unsigned short* __restrict__ protoA,
                                                  const unsigned short* __restrict__ h2,
                                                  const float* __restrict__ x2ps,
                                                  const float* __restrict__ p2,
                                                  unsigned int* __restrict__ mind) {
  int wid  = (blockIdx.x * blockDim.x + threadIdx.x) >> 5;
  int lane = threadIdx.x & 31;
  int mblk = wid & 3;            // 4 M-blocks of 64 (PP=256)
  int rest = wid >> 2;
  int nblk = rest & 31;          // 32 N-blocks of 64 (X2PSN=2048 cols)
  int bb   = rest >> 5;          // 64 batches
  int m0 = mblk * 64, l0 = nblk * 64;
  int m = lane & 15, hi = lane >> 4;

  f32x8 acc[4][4];
#pragma unroll
  for (int mi = 0; mi < 4; ++mi)
#pragma unroll
    for (int ni = 0; ni < 4; ++ni)
      acc[mi][ni] = (f32x8){0,0,0,0,0,0,0,0};

  const unsigned short* arow = protoA + (size_t)(m0 + m) * KPROTO;

  for (int kt = 0; kt < 128; ++kt) {
    int kk = kt >> 2;
    int c0 = (kt & 3) << 5;
    AB16 a[4];
#pragma unroll
    for (int mi = 0; mi < 4; ++mi) {
      const unsigned short* ap = arow + (size_t)mi * 16 * KPROTO + kt * 32;
      a[mi].h[0] = *(const bf16x8*)(ap + 16 * hi);
      a[mi].h[1] = *(const bf16x8*)(ap + 8 + 16 * hi);
    }
    const unsigned short* bp =
        h2 + ((size_t)bb * L1P + (l0 + m + kk)) * LAT + c0 + 16 * hi;
    bf16x16 bfrag[4];
#pragma unroll
    for (int ni = 0; ni < 4; ++ni)
      bfrag[ni] = *(const bf16x16*)(bp + (size_t)ni * 16 * LAT);
#pragma unroll
    for (int ni = 0; ni < 4; ++ni)
#pragma unroll
      for (int mi = 0; mi < 4; ++mi)
        acc[mi][ni] = WMMA_BF16(a[mi].v, bfrag[ni], acc[mi][ni]);
  }

  const float INF = __uint_as_float(0x7F800000u);
#pragma unroll
  for (int mi = 0; mi < 4; ++mi) {
#pragma unroll
    for (int ni = 0; ni < 4; ++ni) {
      int pm = m0 + mi * 16 + 8 * hi;
      int l  = l0 + ni * 16 + m;
      float xps = x2ps[(size_t)bb * X2PSN + l];
      const f32x8& c = acc[mi][ni];
#pragma unroll
      for (int v = 0; v < 8; ++v) {
        int p = pm + v;
        float d = xps - 2.0f * c[v] + p2[p];
        d = d > 0.0f ? d : 0.0f;
        if (l >= LOUT) d = INF;
        // min over the 16 time-columns held by lanes (n = lane&15)
        d = fminf(d, __shfl_xor(d, 1));
        d = fminf(d, __shfl_xor(d, 2));
        d = fminf(d, __shfl_xor(d, 4));
        d = fminf(d, __shfl_xor(d, 8));
        if (m == 0) atomicMin(&mind[(size_t)bb * PP + p], __float_as_uint(d));
      }
    }
  }
}

// ---------------- final: similarity log + classifier + output ----------------
__global__ __launch_bounds__(256) void final_kernel(const unsigned int* __restrict__ mind,
                                                    const float* __restrict__ last_w,
                                                    float* __restrict__ out) {
  int idx = blockIdx.x * blockDim.x + threadIdx.x;
  if (idx < NB * NP) {
    int p = idx % NP, b = idx / NP;
    out[NB * NCLS + idx] = __uint_as_float(mind[(size_t)b * PP + p]);
  }
  if (idx < NB * NCLS) {
    int cls = idx % NCLS, b = idx / NCLS;
    float s = 0.0f;
    for (int p = 0; p < NP; ++p) {
      float md  = __uint_as_float(mind[(size_t)b * PP + p]);
      float act = logf((md + 1.0f) / (md + EPS_));
      s += act * last_w[cls * NP + p];
    }
    out[idx] = s;
  }
}

// ---------------- host launcher ----------------
extern "C" void kernel_launch(void* const* d_in, const int* in_sizes, int n_in,
                              void* d_out, int out_size, void* d_ws, size_t ws_size,
                              hipStream_t stream) {
  const float* x       = (const float*)d_in[0];
  const float* enc_w   = (const float*)d_in[1];
  const float* addon_w = (const float*)d_in[2];
  const float* proto   = (const float*)d_in[3];
  const float* last_w  = (const float*)d_in[4];
  float* out = (float*)d_out;

  unsigned char* ws = (unsigned char*)d_ws;
  size_t off = 0;
  auto alloc = [&](size_t bytes) -> void* {
    void* p = ws + off;
    off = (off + bytes + 255) & ~(size_t)255;
    return p;
  };
  unsigned short* xT     = (unsigned short*)alloc((size_t)NB * TP * FIN * 2);
  unsigned short* encA   = (unsigned short*)alloc((size_t)LAT * KENC * 2);
  unsigned short* addA   = (unsigned short*)alloc((size_t)LAT * LAT * 2);
  unsigned short* protoA = (unsigned short*)alloc((size_t)PP * KPROTO * 2);
  unsigned short* h1     = (unsigned short*)alloc((size_t)NB * L1P * LAT * 2);
  unsigned short* h2     = (unsigned short*)alloc((size_t)NB * L1P * LAT * 2);
  float*          x2s    = (float*)alloc((size_t)NB * L1P * 4);
  float*          x2ps   = (float*)alloc((size_t)NB * X2PSN * 4);
  float*          p2     = (float*)alloc((size_t)PP * 4);
  unsigned int*   mind   = (unsigned int*)alloc((size_t)NB * PP * 4);

  hipLaunchKernelGGL(init_mind, dim3((NB * PP + 255) / 256), dim3(256), 0, stream, mind);
  hipLaunchKernelGGL(pack_xT, dim3((NB * TP * FIN + 255) / 256), dim3(256), 0, stream, x, xT);
  hipLaunchKernelGGL(pack_encA, dim3((LAT * KENC + 255) / 256), dim3(256), 0, stream, enc_w, encA);
  hipLaunchKernelGGL(pack_addonA, dim3((LAT * LAT + 255) / 256), dim3(256), 0, stream, addon_w, addA);
  hipLaunchKernelGGL(pack_protoA, dim3((PP * KPROTO + 255) / 256), dim3(256), 0, stream, proto, protoA);
  hipLaunchKernelGGL(p2_kernel, dim3(1), dim3(256), 0, stream, protoA, p2);

  // 64 b * 65 nblk * 4 mblk = 16640 waves -> 2080 blocks of 8 waves
  hipLaunchKernelGGL(enc_gemm, dim3(2080), dim3(256), 0, stream, encA, xT, h1);
  hipLaunchKernelGGL(addon_gemm, dim3(2080), dim3(256), 0, stream, addA, h1, h2);

  hipLaunchKernelGGL(x2s_kernel, dim3((NB * L1P + 255) / 256), dim3(256), 0, stream, h2, x2s);
  hipLaunchKernelGGL(x2ps_kernel, dim3((NB * X2PSN + 255) / 256), dim3(256), 0, stream, x2s, x2ps);

  // 64 b * 32 nblk * 4 mblk = 8192 waves -> 1024 blocks of 8 waves
  hipLaunchKernelGGL(proto_gemm, dim3(1024), dim3(256), 0, stream,
                     protoA, h2, x2ps, p2, mind);

  hipLaunchKernelGGL(final_kernel, dim3((NB * NP + 255) / 256), dim3(256), 0, stream,
                     mind, last_w, out);
}
